// NodeEarlyInteractionBaseline_9981503995969
// MI455X (gfx1250) — compile-verified, matrix operand order checked by reading, not run on hardware
//
#include <hip/hip_runtime.h>
#include <hip/hip_bf16.h>

typedef __attribute__((ext_vector_type(16))) _Float16 v16h;
typedef __attribute__((ext_vector_type(8)))  float    v8f;

union V16 { v16h v; _Float16 h[16]; };

#define B_       256
#define S_       128
#define D_       64
#define NNODES   (2*B_*S_)     /* 65536 */
#define NEDGES   (2*B_*1024)   /* 524288 */
#define NPROP_   5
#define SINKIT_  20
#define INV_TEMP 10.0f

// ---------------------------------------------------------------- utilities
__global__ __launch_bounds__(256) void zero_kernel(float* __restrict__ p, int n) {
  int i = blockIdx.x * 256 + threadIdx.x;
  int stride = gridDim.x * 256;
  for (; i < n; i += stride) p[i] = 0.f;
}

// ------------------------------------------------------------ input encoders
template<int KIN, int DOUT>
__global__ __launch_bounds__(256) void enc_kernel(const float* __restrict__ feat,
                                                  const float* __restrict__ W,
                                                  const float* __restrict__ bias,
                                                  float* __restrict__ out, int nrows) {
  __shared__ float sW[KIN * DOUT];
  __shared__ float sb[DOUT];
  for (int i = threadIdx.x; i < KIN * DOUT; i += 256) sW[i] = W[i];
  for (int i = threadIdx.x; i < DOUT; i += 256) sb[i] = bias[i];
  __syncthreads();
  int gid = blockIdx.x * 256 + threadIdx.x;
  if (gid >= nrows * DOUT) return;
  int r = gid / DOUT, d = gid % DOUT;
  float s = sb[d];
  const float* f = feat + (long)r * KIN;
#pragma unroll
  for (int k = 0; k < KIN; ++k) s += f[k] * sW[k * DOUT + d];
  out[gid] = s;
}

// ------------------------------------------------- generic 2-layer node MLP
// out = relu([src0|src1] @ W1 + b1) @ W2 + b2 , rows = 128 per block
// Weights staged in LDS as f16, transposed so B-fragments are contiguous
// 16B runs (coalesced global reads; transpose happens on the LDS-write side).
template<int K1, int H, int OUT>
__global__ __launch_bounds__(256) void mlp2_kernel(
    const float* __restrict__ src0, int w0,
    const float* __restrict__ src1, int w1,
    const float* __restrict__ W1, const float* __restrict__ b1,
    const float* __restrict__ W2, const float* __restrict__ b2,
    float* __restrict__ out) {
  __shared__ _Float16 sW1[H * K1];   // transposed: sW1[n*K1+k]
  __shared__ _Float16 sW2[OUT * H];  // transposed: sW2[n*H +k]
  __shared__ float sb1[H];
  __shared__ float sb2[OUT];
  __shared__ _Float16 sY[8 * 16 * H];

  int tid = threadIdx.x;
  for (int i = tid; i < K1 * H; i += 256) {      // linear (coalesced) global read
    int k = i / H, n = i % H;                    // H is power of two -> shifts
    sW1[n * K1 + k] = (_Float16)W1[i];
  }
  for (int i = tid; i < H * OUT; i += 256) {
    int k = i / OUT, n = i % OUT;
    sW2[n * H + k] = (_Float16)W2[i];
  }
  for (int i = tid; i < H;   i += 256) sb1[i] = b1[i];
  for (int i = tid; i < OUT; i += 256) sb2[i] = b2[i];
  __syncthreads();

  const int lane = tid & 31, wave = tid >> 5;
  const int m  = lane & 15;
  const int kb = (lane >> 4) * 8;
  const int mo = kb;
  const long r = (long)blockIdx.x * 128 + wave * 16 + m;

  constexpr int NC1 = K1 / 32;
  V16 a[NC1];
#pragma unroll
  for (int s = 0; s < NC1; ++s) {
#pragma unroll
    for (int hs = 0; hs < 2; ++hs) {
      int ks = 32 * s + hs * 16 + kb;
      const float* p = (ks < w0) ? (src0 + r * w0 + ks) : (src1 + r * w1 + (ks - w0));
#pragma unroll
      for (int i = 0; i < 8; ++i) a[s].h[hs * 8 + i] = (_Float16)p[i];
    }
  }

  _Float16* yw = sY + wave * 16 * H;
#pragma unroll
  for (int t = 0; t < H / 16; ++t) {
    v8f acc = {};
#pragma unroll
    for (int s = 0; s < NC1; ++s) {
      V16 bf;
      const _Float16* wp = sW1 + (t * 16 + m) * K1 + 32 * s;
#pragma unroll
      for (int i = 0; i < 8; ++i) { bf.h[i] = wp[kb + i]; bf.h[8 + i] = wp[16 + kb + i]; }
      acc = __builtin_amdgcn_wmma_f32_16x16x32_f16(false, a[s].v, false, bf.v, (short)0, acc, false, false);
    }
    int col = t * 16 + m;
    float bias = sb1[col];
#pragma unroll
    for (int j = 0; j < 8; ++j) {
      float y = acc[j] + bias;
      yw[(mo + j) * H + col] = (_Float16)(y > 0.f ? y : 0.f);
    }
  }

  constexpr int NC2 = H / 32;
  V16 ay[NC2];
#pragma unroll
  for (int s = 0; s < NC2; ++s) {
    const _Float16* yp = yw + m * H + 32 * s;
#pragma unroll
    for (int i = 0; i < 8; ++i) { ay[s].h[i] = yp[kb + i]; ay[s].h[8 + i] = yp[16 + kb + i]; }
  }

#pragma unroll
  for (int t = 0; t < OUT / 16; ++t) {
    v8f acc = {};
#pragma unroll
    for (int s = 0; s < NC2; ++s) {
      V16 bf;
      const _Float16* wp = sW2 + (t * 16 + m) * H + 32 * s;
#pragma unroll
      for (int i = 0; i < 8; ++i) { bf.h[i] = wp[kb + i]; bf.h[8 + i] = wp[16 + kb + i]; }
      acc = __builtin_amdgcn_wmma_f32_16x16x32_f16(false, ay[s].v, false, bf.v, (short)0, acc, false, false);
    }
    int col = t * 16 + m;
    float bias = sb2[col];
    long rb = (long)blockIdx.x * 128 + wave * 16 + mo;
#pragma unroll
    for (int j = 0; j < 8; ++j) out[(rb + j) * OUT + col] = acc[j] + bias;
  }
}

// -------------------------------------------------- edge message MLP + scatter
// for each edge: m_f = relu([hf|ht|ee]@W1+b1)@W2+b2 -> atomicAdd agg[to]
//                m_r = relu([ht|hf|ee]@W1+b1)@W2+b2 -> atomicAdd agg[from]
// Reverse direction reuses the forward A-fragments via K-chunk permutation.
__global__ __launch_bounds__(256) void msg_kernel(
    const float* __restrict__ comb, const float* __restrict__ e_enc,
    const int* __restrict__ from_idx, const int* __restrict__ to_idx,
    const float* __restrict__ W1, const float* __restrict__ b1,
    const float* __restrict__ W2, const float* __restrict__ b2,
    float* __restrict__ agg) {
  constexpr int K1 = 160, H = 128, OUT = 128;
  __shared__ _Float16 sW1[H * K1];
  __shared__ _Float16 sW2[OUT * H];
  __shared__ float sb1[H];
  __shared__ float sb2[OUT];
  __shared__ _Float16 sY[8 * 16 * H];

  int tid = threadIdx.x;
  for (int i = tid; i < K1 * H; i += 256) {      // coalesced read, transposed LDS write
    int k = i >> 7, n = i & 127;                 // H = 128
    sW1[n * K1 + k] = (_Float16)W1[i];
  }
  for (int i = tid; i < H * OUT; i += 256) {
    int k = i >> 7, n = i & 127;                 // OUT = 128
    sW2[n * H + k] = (_Float16)W2[i];
  }
  for (int i = tid; i < H; i += 256) { sb1[i] = b1[i]; sb2[i] = b2[i]; }
  __syncthreads();

  const int lane = tid & 31, wave = tid >> 5;
  const int m  = lane & 15;
  const int kb = (lane >> 4) * 8;
  const int mo = kb;
  const int ebase = blockIdx.x * 128 + wave * 16;
  const int e  = ebase + m;
  const int fi = from_idx[e], ti = to_idx[e];

  int tof[8], fro[8];
#pragma unroll
  for (int j = 0; j < 8; ++j) { tof[j] = to_idx[ebase + mo + j]; fro[j] = from_idx[ebase + mo + j]; }

  V16 a[5];  // x = [hf(0:64) | ht(64:128) | ee(128:160)]
#pragma unroll
  for (int s = 0; s < 5; ++s) {
#pragma unroll
    for (int hs = 0; hs < 2; ++hs) {
      int ks = 32 * s + hs * 16 + kb;
      const float* p = (ks < 64)  ? (comb + (long)fi * 64 + ks)
                     : (ks < 128) ? (comb + (long)ti * 64 + (ks - 64))
                                  : (e_enc + (long)e * 32 + (ks - 128));
#pragma unroll
      for (int i = 0; i < 8; ++i) a[s].h[hs * 8 + i] = (_Float16)p[i];
    }
  }

  _Float16* yw = sY + wave * 16 * H;
#pragma unroll
  for (int dir = 0; dir < 2; ++dir) {
#pragma unroll
    for (int t = 0; t < H / 16; ++t) {
      v8f acc = {};
#pragma unroll
      for (int s = 0; s < 5; ++s) {
        int as = (dir == 0) ? s : (s < 2 ? s + 2 : (s < 4 ? s - 2 : 4)); // swap hf<->ht
        V16 bf;
        const _Float16* wp = sW1 + (t * 16 + m) * K1 + 32 * s;
#pragma unroll
        for (int i = 0; i < 8; ++i) { bf.h[i] = wp[kb + i]; bf.h[8 + i] = wp[16 + kb + i]; }
        acc = __builtin_amdgcn_wmma_f32_16x16x32_f16(false, a[as].v, false, bf.v, (short)0, acc, false, false);
      }
      int col = t * 16 + m;
      float bias = sb1[col];
#pragma unroll
      for (int j = 0; j < 8; ++j) {
        float y = acc[j] + bias;
        yw[(mo + j) * H + col] = (_Float16)(y > 0.f ? y : 0.f);
      }
    }

    V16 ay[4];
#pragma unroll
    for (int s = 0; s < 4; ++s) {
      const _Float16* yp = yw + m * H + 32 * s;
#pragma unroll
      for (int i = 0; i < 8; ++i) { ay[s].h[i] = yp[kb + i]; ay[s].h[8 + i] = yp[16 + kb + i]; }
    }

#pragma unroll
    for (int t = 0; t < OUT / 16; ++t) {
      v8f acc = {};
#pragma unroll
      for (int s = 0; s < 4; ++s) {
        V16 bf;
        const _Float16* wp = sW2 + (t * 16 + m) * H + 32 * s;
#pragma unroll
        for (int i = 0; i < 8; ++i) { bf.h[i] = wp[kb + i]; bf.h[8 + i] = wp[16 + kb + i]; }
        acc = __builtin_amdgcn_wmma_f32_16x16x32_f16(false, ay[s].v, false, bf.v, (short)0, acc, false, false);
      }
      int col = t * 16 + m;
      float bias = sb2[col];
#pragma unroll
      for (int j = 0; j < 8; ++j) {
        int dst = dir ? fro[j] : tof[j];
        atomicAdd(&agg[(long)dst * 128 + col], acc[j] + bias);
      }
    }
  }
}

// ------------------------------------------------------------- sim = tq @ tc^T
__global__ __launch_bounds__(256) void sim_kernel(const float* __restrict__ tnode,
                                                  float* __restrict__ T) {
  int tid = threadIdx.x, b = blockIdx.x;
  int lane = tid & 31, wave = tid >> 5;
  int m = lane & 15, kb = (lane >> 4) * 8, mo = kb;
  long qnode = (long)(2 * b) * S_ + wave * 16 + m;
  V16 aq;
#pragma unroll
  for (int hs = 0; hs < 2; ++hs) {
    const float* p = tnode + qnode * 32 + hs * 16 + kb;
#pragma unroll
    for (int i = 0; i < 8; ++i) aq.h[hs * 8 + i] = (_Float16)p[i];
  }
#pragma unroll
  for (int t = 0; t < 8; ++t) {
    long cnode = (long)(2 * b + 1) * S_ + t * 16 + m;
    V16 bf;
#pragma unroll
    for (int hs = 0; hs < 2; ++hs) {
      const float* p = tnode + cnode * 32 + hs * 16 + kb;
#pragma unroll
      for (int i = 0; i < 8; ++i) bf.h[hs * 8 + i] = (_Float16)p[i];
    }
    v8f acc = {};
    acc = __builtin_amdgcn_wmma_f32_16x16x32_f16(false, aq.v, false, bf.v, (short)0, acc, false, false);
#pragma unroll
    for (int j = 0; j < 8; ++j)
      T[(long)b * 16384 + (wave * 16 + mo + j) * 128 + t * 16 + m] = acc[j];
  }
}

// ------------------------------------------------------ sinkhorn (in-LDS)
__global__ __launch_bounds__(256) void sinkhorn_kernel(float* __restrict__ T) {
  __shared__ float M[128 * 129];   // padded stride kills bank conflicts
  int tid = threadIdx.x, b = blockIdx.x;
  float* Tb = T + (long)b * 16384;
  for (int i = tid; i < 16384; i += 256) M[(i >> 7) * 129 + (i & 127)] = Tb[i] * INV_TEMP;
  __syncthreads();
  for (int it = 0; it < SINKIT_; ++it) {
    if (tid < 128) {                     // axis=2 (rows over j)
      float* row = M + tid * 129;
      float mx = -1e30f;
      for (int j = 0; j < 128; ++j) mx = fmaxf(mx, row[j]);
      float s = 0.f;
      for (int j = 0; j < 128; ++j) s += __expf(row[j] - mx);
      float lse = mx + __logf(s);
      for (int j = 0; j < 128; ++j) row[j] -= lse;
    }
    __syncthreads();
    if (tid < 128) {                     // axis=1 (cols over i)
      float mx = -1e30f;
      for (int i = 0; i < 128; ++i) mx = fmaxf(mx, M[i * 129 + tid]);
      float s = 0.f;
      for (int i = 0; i < 128; ++i) s += __expf(M[i * 129 + tid] - mx);
      float lse = mx + __logf(s);
      for (int i = 0; i < 128; ++i) M[i * 129 + tid] -= lse;
    }
    __syncthreads();
  }
  for (int i = tid; i < 16384; i += 256) Tb[i] = __expf(M[(i >> 7) * 129 + (i & 127)]);
}

// ------------------------------- store = interleave(T@c, T^T@q) per batch
__global__ __launch_bounds__(256) void match_kernel(const float* __restrict__ T,
                                                    const float* __restrict__ h,
                                                    float* __restrict__ store) {
  __shared__ _Float16 sT [128 * 128];
  __shared__ _Float16 sTt[128 * 128];
  __shared__ _Float16 sCt[64 * 128];
  __shared__ _Float16 sQt[64 * 128];
  int tid = threadIdx.x, b = blockIdx.x;
  const float* Tb = T + (long)b * 16384;
  for (int idx = tid; idx < 16384; idx += 256) {
    int i = idx >> 7, j = idx & 127;
    _Float16 v = (_Float16)Tb[idx];
    sT[idx] = v;
    sTt[j * 128 + i] = v;
  }
  long qbase = (long)(2 * b) * S_, cbase = (long)(2 * b + 1) * S_;
  for (int idx = tid; idx < 8192; idx += 256) {
    int j = idx >> 6, d = idx & 63;
    sQt[d * 128 + j] = (_Float16)h[(qbase + j) * 64 + d];
    sCt[d * 128 + j] = (_Float16)h[(cbase + j) * 64 + d];
  }
  __syncthreads();

  int lane = tid & 31, wave = tid >> 5;
  int m = lane & 15, kb = (lane >> 4) * 8, mo = kb;

  // q_from_c[i][d] = sum_j T[i][j] c[j][d]
  {
    V16 a[4];
#pragma unroll
    for (int s = 0; s < 4; ++s) {
      const _Float16* p = sT + (wave * 16 + m) * 128 + 32 * s;
#pragma unroll
      for (int i = 0; i < 8; ++i) { a[s].h[i] = p[kb + i]; a[s].h[8 + i] = p[16 + kb + i]; }
    }
#pragma unroll
    for (int t = 0; t < 4; ++t) {
      v8f acc = {};
#pragma unroll
      for (int s = 0; s < 4; ++s) {
        V16 bf;
        const _Float16* p = sCt + (t * 16 + m) * 128 + 32 * s;
#pragma unroll
        for (int i = 0; i < 8; ++i) { bf.h[i] = p[kb + i]; bf.h[8 + i] = p[16 + kb + i]; }
        acc = __builtin_amdgcn_wmma_f32_16x16x32_f16(false, a[s].v, false, bf.v, (short)0, acc, false, false);
      }
#pragma unroll
      for (int j = 0; j < 8; ++j)
        store[(qbase + wave * 16 + mo + j) * 64 + t * 16 + m] = acc[j];
    }
  }
  // c_from_q[j][d] = sum_i T[i][j] q[i][d]
  {
    V16 a[4];
#pragma unroll
    for (int s = 0; s < 4; ++s) {
      const _Float16* p = sTt + (wave * 16 + m) * 128 + 32 * s;
#pragma unroll
      for (int i = 0; i < 8; ++i) { a[s].h[i] = p[kb + i]; a[s].h[8 + i] = p[16 + kb + i]; }
    }
#pragma unroll
    for (int t = 0; t < 4; ++t) {
      v8f acc = {};
#pragma unroll
      for (int s = 0; s < 4; ++s) {
        V16 bf;
        const _Float16* p = sQt + (t * 16 + m) * 128 + 32 * s;
#pragma unroll
        for (int i = 0; i < 8; ++i) { bf.h[i] = p[kb + i]; bf.h[8 + i] = p[16 + kb + i]; }
        acc = __builtin_amdgcn_wmma_f32_16x16x32_f16(false, a[s].v, false, bf.v, (short)0, acc, false, false);
      }
#pragma unroll
      for (int j = 0; j < 8; ++j)
        store[(cbase + wave * 16 + mo + j) * 64 + t * 16 + m] = acc[j];
    }
  }
}

// ---------------------------------------------- scores[b] = -sum relu(q - T@c)
__global__ __launch_bounds__(256) void score_kernel(const float* __restrict__ h,
                                                    const float* __restrict__ store,
                                                    float* __restrict__ out) {
  __shared__ float red[256];
  int tid = threadIdx.x, b = blockIdx.x;
  long base = (long)b * 16384;   // q-rows of batch b are contiguous
  float s = 0.f;
  for (int i = tid; i < 8192; i += 256) {
    float v = h[base + i] - store[base + i];
    s += (v > 0.f ? v : 0.f);
  }
  red[tid] = s;
  __syncthreads();
  for (int off = 128; off > 0; off >>= 1) {
    if (tid < off) red[tid] += red[tid + off];
    __syncthreads();
  }
  if (tid == 0) out[b] = -red[0];
}

// --------------------------------------------------------------------- host
extern "C" void kernel_launch(void* const* d_in, const int* in_sizes, int n_in,
                              void* d_out, int out_size, void* d_ws, size_t ws_size,
                              hipStream_t stream) {
  (void)in_sizes; (void)n_in; (void)out_size; (void)ws_size;
  const float* node_features = (const float*)d_in[0];
  const float* edge_features = (const float*)d_in[1];
  const float* W_enc_n = (const float*)d_in[2];  const float* b_enc_n = (const float*)d_in[3];
  const float* W_enc_e = (const float*)d_in[4];  const float* b_enc_e = (const float*)d_in[5];
  const float* W_c1 = (const float*)d_in[6];  const float* b_c1 = (const float*)d_in[7];
  const float* W_c2 = (const float*)d_in[8];  const float* b_c2 = (const float*)d_in[9];
  const float* W_m1 = (const float*)d_in[10]; const float* b_m1 = (const float*)d_in[11];
  const float* W_m2 = (const float*)d_in[12]; const float* b_m2 = (const float*)d_in[13];
  const float* W_u1 = (const float*)d_in[14]; const float* b_u1 = (const float*)d_in[15];
  const float* W_u2 = (const float*)d_in[16]; const float* b_u2 = (const float*)d_in[17];
  const float* W_t1 = (const float*)d_in[18]; const float* b_t1 = (const float*)d_in[19];
  const float* W_t2 = (const float*)d_in[20]; const float* b_t2 = (const float*)d_in[21];
  const int* from_idx = (const int*)d_in[22];
  const int* to_idx   = (const int*)d_in[23];

  float* ws    = (float*)d_ws;
  float* e_enc = ws;  ws += (size_t)NEDGES * 32;
  float* h     = ws;  ws += (size_t)NNODES * 64;
  float* store = ws;  ws += (size_t)NNODES * 64;
  float* comb  = ws;  ws += (size_t)NNODES * 64;
  float* agg   = ws;  ws += (size_t)NNODES * 128;
  float* tnode = ws;  ws += (size_t)NNODES * 32;
  float* Tbuf  = ws;  ws += (size_t)B_ * S_ * S_;

  zero_kernel<<<1024, 256, 0, stream>>>(store, NNODES * 64);
  enc_kernel<16, 64><<<NNODES * 64 / 256, 256, 0, stream>>>(node_features, W_enc_n, b_enc_n, h, NNODES);
  enc_kernel<8, 32><<<NEDGES * 32 / 256, 256, 0, stream>>>(edge_features, W_enc_e, b_enc_e, e_enc, NEDGES);

  for (int p = 0; p < NPROP_; ++p) {
    mlp2_kernel<128, 128, 64><<<NNODES / 128, 256, 0, stream>>>(h, 64, store, 64, W_c1, b_c1, W_c2, b_c2, comb);
    zero_kernel<<<2048, 256, 0, stream>>>(agg, NNODES * 128);
    msg_kernel<<<NEDGES / 128, 256, 0, stream>>>(comb, e_enc, from_idx, to_idx, W_m1, b_m1, W_m2, b_m2, agg);
    mlp2_kernel<192, 128, 64><<<NNODES / 128, 256, 0, stream>>>(comb, 64, agg, 128, W_u1, b_u1, W_u2, b_u2, h);
    mlp2_kernel<64, 32, 32><<<NNODES / 128, 256, 0, stream>>>(h, 64, h, 0, W_t1, b_t1, W_t2, b_t2, tnode);
    sim_kernel<<<B_, 256, 0, stream>>>(tnode, Tbuf);
    sinkhorn_kernel<<<B_, 256, 0, stream>>>(Tbuf);
    match_kernel<<<B_, 256, 0, stream>>>(Tbuf, h, store);
  }
  score_kernel<<<B_, 256, 0, stream>>>(h, store, (float*)d_out);
}
